// GraphAttentionLayer_51213190038008
// MI455X (gfx1250) — compile-verified, matrix operand order checked by reading, not run on hardware
//
#include <hip/hip_runtime.h>

#define S_DIM 8192
#define R_DIM 8192
#define N_DIM 4096
#define F_DIM 64
#define LRELU_A 0.2f
#define BN_EPS 1e-5f

typedef __bf16 bf16;
typedef __attribute__((ext_vector_type(16))) __bf16 v16bf;
typedef __attribute__((ext_vector_type(8)))  __bf16 v8bf;
typedef __attribute__((ext_vector_type(8)))  float  v8f;

__device__ __forceinline__ float lrelu(float x) { return x > 0.f ? x : LRELU_A * x; }

// Build a 16x32 bf16 WMMA fragment from memory where the 16 "rows" are
// contiguous-in-K (row-major A, or N-major/transposed B). Two b128 loads/lane.
// ISA layout: lane L -> row L&15; VGPR0..3 hold K = 8g+0..7 (pairs), VGPR4..7
// hold K = 16+8g+0..7, where g = L>>4.
__device__ __forceinline__ v16bf load_frag(const bf16* __restrict__ base, long ld, int lane) {
  const int row = lane & 15, g = lane >> 4;
  const bf16* p = base + (long)row * ld + g * 8;
  v8bf lo = *(const v8bf*)(p);        // K = 8g .. 8g+7
  v8bf hi = *(const v8bf*)(p + 16);   // K = 16+8g .. 16+8g+7
  v16bf r;
#pragma unroll
  for (int e = 0; e < 8; ++e) { r[e] = lo[e]; r[8 + e] = hi[e]; }
  return r;
}

// ---------------- small projection kernels ----------------

// h1[j,:] = recip[j,:] @ W1 ; r1[j] = h1[j,:] . a12[:64] ; store h1T bf16 (64 x 8192)
__global__ void k_h1(const float* __restrict__ recip, const float* __restrict__ W1,
                     const float* __restrict__ a12, bf16* __restrict__ h1T,
                     float* __restrict__ r1) {
  const int j = blockIdx.x, t = threadIdx.x;            // 64 threads
  __shared__ float row[F_DIM];
  __shared__ float red[F_DIM];
  row[t] = recip[(long)j * F_DIM + t];
  __syncthreads();
  float acc = 0.f;
#pragma unroll
  for (int k = 0; k < F_DIM; ++k) acc += row[k] * W1[k * F_DIM + t];
  h1T[(long)t * R_DIM + j] = (bf16)acc;
  red[t] = acc * a12[t];
  __syncthreads();
  for (int s = 32; s > 0; s >>= 1) { if (t < s) red[t] += red[t + s]; __syncthreads(); }
  if (t == 0) r1[j] = red[0];
}

// src = gather(source_embedding); h2 = src @ W2; s2/l3/r3 dots; srcT/h2T bf16 (64 x 4096)
__global__ void k_h2src(const float* __restrict__ semb, const int* __restrict__ sidx,
                        const float* __restrict__ W2, const float* __restrict__ a12,
                        const float* __restrict__ a3, bf16* __restrict__ h2T,
                        bf16* __restrict__ srcT, float* __restrict__ s2,
                        float* __restrict__ l3, float* __restrict__ r3) {
  const int i = blockIdx.x, t = threadIdx.x;            // 64 threads
  __shared__ float row[F_DIM];
  __shared__ float red[F_DIM];
  const int si = sidx[i];
  row[t] = semb[(long)si * F_DIM + t];
  __syncthreads();
  float acc = 0.f;
#pragma unroll
  for (int k = 0; k < F_DIM; ++k) acc += row[k] * W2[k * F_DIM + t];
  h2T[(long)t * N_DIM + i]  = (bf16)acc;
  srcT[(long)t * N_DIM + i] = (bf16)row[t];
  red[t] = acc * a12[F_DIM + t];
  __syncthreads();
  for (int s = 32; s > 0; s >>= 1) { if (t < s) red[t] += red[t + s]; __syncthreads(); }
  if (t == 0) s2[i] = red[0];
  __syncthreads();
  red[t] = acc * a3[t];
  __syncthreads();
  for (int s = 32; s > 0; s >>= 1) { if (t < s) red[t] += red[t + s]; __syncthreads(); }
  if (t == 0) l3[i] = red[0];
  __syncthreads();
  red[t] = acc * a3[F_DIM + t];
  __syncthreads();
  for (int s = 32; s > 0; s >>= 1) { if (t < s) red[t] += red[t + s]; __syncthreads(); }
  if (t == 0) r3[i] = red[0];
}

// ---------------- masked-softmax row sums ----------------

__global__ void k_sums(const float* __restrict__ adjI, const float* __restrict__ adjA,
                       const int* __restrict__ sidx, const float* __restrict__ s2,
                       const float* __restrict__ r1, const float* __restrict__ l3,
                       const float* __restrict__ r3, float* __restrict__ invI,
                       float* __restrict__ invC) {
  const int i = blockIdx.x, t = threadIdx.x;            // 256 threads
  const int si = sidx[i];
  const float s2i = s2[i], l3i = l3[i];
  __shared__ float red[256];
  __shared__ float SI_sh;
  float sI = 0.f;
  for (int j = t; j < R_DIM; j += 256)
    if (adjI[(long)si * R_DIM + j] > 0.f) sI += expf(lrelu(s2i + r1[j]));
  red[t] = sI; __syncthreads();
  for (int s = 128; s > 0; s >>= 1) { if (t < s) red[t] += red[t + s]; __syncthreads(); }
  if (t == 0) { SI_sh = red[0]; invI[i] = 1.f / red[0]; }
  __syncthreads();
  float sA = 0.f;
  for (int j = t; j < N_DIM; j += 256)
    if (adjA[(long)si * S_DIM + sidx[j]] > 0.f) sA += expf(lrelu(l3i + r3[j]));
  red[t] = sA; __syncthreads();
  for (int s = 128; s > 0; s >>= 1) { if (t < s) red[t] += red[t + s]; __syncthreads(); }
  if (t == 0) invC[i] = 1.f / (SI_sh + red[0]);
}

// ---------------- materialize normalized attention (bf16) ----------------

// attI (4096 x 8192) row-major + attT (8192 x 4096) transposed copy
__global__ void k_att_inter(const float* __restrict__ adjI, const int* __restrict__ sidx,
                            const float* __restrict__ s2, const float* __restrict__ r1,
                            const float* __restrict__ invI, bf16* __restrict__ attI,
                            bf16* __restrict__ attT) {
  const int j  = blockIdx.x * 256 + threadIdx.x;
  const int i0 = blockIdx.y * 16;
  const float r1j = r1[j];
#pragma unroll
  for (int ii = 0; ii < 16; ++ii) {
    const int i = i0 + ii;
    float v = 0.f;
    if (adjI[(long)sidx[i] * R_DIM + j] > 0.f)
      v = expf(lrelu(s2[i] + r1j)) * invI[i];
    const bf16 b = (bf16)v;
    attI[(long)i * R_DIM + j] = b;            // coalesced across threads
    attT[(long)j * N_DIM + i] = b;            // 32B contiguous per thread
  }
}

// attA (4096 x 4096) row-major only
__global__ void k_att_intra(const float* __restrict__ adjA, const int* __restrict__ sidx,
                            const float* __restrict__ l3, const float* __restrict__ r3,
                            const float* __restrict__ invC, bf16* __restrict__ attA) {
  const int j  = blockIdx.x * 256 + threadIdx.x;
  const int i0 = blockIdx.y * 16;
  const int sj = sidx[j];
  const float r3j = r3[j];
#pragma unroll
  for (int ii = 0; ii < 16; ++ii) {
    const int i = i0 + ii;
    float v = 0.f;
    if (adjA[(long)sidx[i] * S_DIM + sj] > 0.f)
      v = expf(lrelu(l3[i] + r3j)) * invC[i];
    attA[(long)i * N_DIM + j] = (bf16)v;
  }
}

// ---------------- WMMA GEMMs ----------------

// u_pre = attI @ h1 + attA @ h2   (4096 x 64). 128 thr = 4 waves, wave w -> cols 16w..
__global__ void k_upre(const bf16* __restrict__ attI, const bf16* __restrict__ attA,
                       const bf16* __restrict__ h1T, const bf16* __restrict__ h2T,
                       float* __restrict__ u_pre) {
  const int i0 = blockIdx.x * 16;
  const int wave = threadIdx.x >> 5, lane = threadIdx.x & 31;
  const int n0 = wave * 16;
  v8f acc = {};
  for (int k0 = 0; k0 < R_DIM; k0 += 32) {
    v16bf a = load_frag(attI + (long)i0 * R_DIM + k0, R_DIM, lane);
    v16bf b = load_frag(h1T  + (long)n0 * R_DIM + k0, R_DIM, lane);
    acc = __builtin_amdgcn_wmma_f32_16x16x32_bf16(false, a, false, b, (short)0, acc, false, false);
  }
  for (int k0 = 0; k0 < N_DIM; k0 += 32) {
    v16bf a = load_frag(attA + (long)i0 * N_DIM + k0, N_DIM, lane);
    v16bf b = load_frag(h2T  + (long)n0 * N_DIM + k0, N_DIM, lane);
    acc = __builtin_amdgcn_wmma_f32_16x16x32_bf16(false, a, false, b, (short)0, acc, false, false);
  }
  const int nn = lane & 15, g = lane >> 4;
#pragma unroll
  for (int r = 0; r < 8; ++r)
    u_pre[(long)(i0 + r + 8 * g) * F_DIM + n0 + nn] = acc[r];
}

// t = attI.T @ src  (8192 x 64) using attT (8192 x 4096) and srcT (64 x 4096)
__global__ void k_t(const bf16* __restrict__ attT, const bf16* __restrict__ srcT,
                    float* __restrict__ tmat) {
  const int j0 = blockIdx.x * 16;
  const int wave = threadIdx.x >> 5, lane = threadIdx.x & 31;
  const int n0 = wave * 16;
  v8f acc = {};
  for (int k0 = 0; k0 < N_DIM; k0 += 32) {
    v16bf a = load_frag(attT + (long)j0 * N_DIM + k0, N_DIM, lane);
    v16bf b = load_frag(srcT + (long)n0 * N_DIM + k0, N_DIM, lane);
    acc = __builtin_amdgcn_wmma_f32_16x16x32_bf16(false, a, false, b, (short)0, acc, false, false);
  }
  const int nn = lane & 15, g = lane >> 4;
#pragma unroll
  for (int r = 0; r < 8; ++r)
    tmat[(long)(j0 + r + 8 * g) * F_DIM + n0 + nn] = acc[r];
}

// v_pre = t @ W1  (8192 x 64)
__global__ void k_vpre(const float* __restrict__ tmat, const float* __restrict__ W1,
                       float* __restrict__ v_pre) {
  const int j = blockIdx.x, t = threadIdx.x;            // 64 threads
  __shared__ float row[F_DIM];
  row[t] = tmat[(long)j * F_DIM + t];
  __syncthreads();
  float acc = 0.f;
#pragma unroll
  for (int k = 0; k < F_DIM; ++k) acc += row[k] * W1[k * F_DIM + t];
  v_pre[(long)j * F_DIM + t] = acc;
}

// column batch-norm (population var) + lrelu, output bf16 row-major
__global__ void k_bn(const float* __restrict__ x, bf16* __restrict__ y, int rows) {
  const int c = blockIdx.x, t = threadIdx.x;            // 256 threads, one column
  __shared__ float rs[256], rss[256];
  __shared__ float mean_sh, inv_sh;
  float s = 0.f, ss = 0.f;
  for (int r = t; r < rows; r += 256) {
    float v = x[(long)r * F_DIM + c];
    s += v; ss += v * v;
  }
  rs[t] = s; rss[t] = ss; __syncthreads();
  for (int k = 128; k > 0; k >>= 1) {
    if (t < k) { rs[t] += rs[t + k]; rss[t] += rss[t + k]; }
    __syncthreads();
  }
  if (t == 0) {
    float m = rs[0] / rows;
    float var = rss[0] / rows - m * m;
    mean_sh = m; inv_sh = rsqrtf(var + BN_EPS);
  }
  __syncthreads();
  const float m = mean_sh, inv = inv_sh;
  for (int r = t; r < rows; r += 256) {
    float v = (x[(long)r * F_DIM + c] - m) * inv;
    y[(long)r * F_DIM + c] = (bf16)lrelu(v);
  }
}

// out = elu(u @ v.T)  (4096 x 8192), K = 64 = two WMMA k-steps
__global__ void k_out(const bf16* __restrict__ u_bf, const bf16* __restrict__ v_bf,
                      float* __restrict__ out) {
  const int i0 = blockIdx.x * 16;
  const int wave = threadIdx.x >> 5, lane = threadIdx.x & 31;
  const int j0 = blockIdx.y * 64 + wave * 16;
  v8f acc = {};
  v16bf a0 = load_frag(u_bf + (long)i0 * F_DIM + 0,  F_DIM, lane);
  v16bf b0 = load_frag(v_bf + (long)j0 * F_DIM + 0,  F_DIM, lane);
  acc = __builtin_amdgcn_wmma_f32_16x16x32_bf16(false, a0, false, b0, (short)0, acc, false, false);
  v16bf a1 = load_frag(u_bf + (long)i0 * F_DIM + 32, F_DIM, lane);
  v16bf b1 = load_frag(v_bf + (long)j0 * F_DIM + 32, F_DIM, lane);
  acc = __builtin_amdgcn_wmma_f32_16x16x32_bf16(false, a1, false, b1, (short)0, acc, false, false);
  const int nn = lane & 15, g = lane >> 4;
#pragma unroll
  for (int r = 0; r < 8; ++r) {
    float x = acc[r];
    out[(long)(i0 + r + 8 * g) * R_DIM + j0 + nn] = x > 0.f ? x : expm1f(x);
  }
}

extern "C" void kernel_launch(void* const* d_in, const int* in_sizes, int n_in,
                              void* d_out, int out_size, void* d_ws, size_t ws_size,
                              hipStream_t stream) {
  (void)in_sizes; (void)n_in; (void)out_size; (void)ws_size;
  const float* adjI = (const float*)d_in[0];
  const float* adjA = (const float*)d_in[1];
  const int*   sidx = (const int*)d_in[2];
  const float* semb = (const float*)d_in[3];
  const float* remb = (const float*)d_in[4];
  const float* W1   = (const float*)d_in[5];
  const float* W2   = (const float*)d_in[6];
  const float* a12  = (const float*)d_in[7];
  const float* a3   = (const float*)d_in[8];
  float* out = (float*)d_out;

  char* w = (char*)d_ws;
  auto carve = [&](size_t bytes) { void* p = (void*)w; w += (bytes + 255) & ~(size_t)255; return p; };

  bf16* attI  = (bf16*)carve((size_t)N_DIM * R_DIM * 2);   // 64 MB
  bf16* attT  = (bf16*)carve((size_t)R_DIM * N_DIM * 2);   // 64 MB
  bf16* attA  = (bf16*)carve((size_t)N_DIM * N_DIM * 2);   // 32 MB
  bf16* h1T   = (bf16*)carve((size_t)F_DIM * R_DIM * 2);
  bf16* h2T   = (bf16*)carve((size_t)F_DIM * N_DIM * 2);
  bf16* srcT  = (bf16*)carve((size_t)F_DIM * N_DIM * 2);
  bf16* u_bf  = (bf16*)carve((size_t)N_DIM * F_DIM * 2);
  bf16* v_bf  = (bf16*)carve((size_t)R_DIM * F_DIM * 2);
  float* u_pre = (float*)carve((size_t)N_DIM * F_DIM * 4);
  float* tmat  = (float*)carve((size_t)R_DIM * F_DIM * 4);
  float* v_pre = (float*)carve((size_t)R_DIM * F_DIM * 4);
  float* s2   = (float*)carve(N_DIM * 4);
  float* l3   = (float*)carve(N_DIM * 4);
  float* r3   = (float*)carve(N_DIM * 4);
  float* r1   = (float*)carve(R_DIM * 4);
  float* invI = (float*)carve(N_DIM * 4);
  float* invC = (float*)carve(N_DIM * 4);

  k_h1<<<R_DIM, F_DIM, 0, stream>>>(remb, W1, a12, h1T, r1);
  k_h2src<<<N_DIM, F_DIM, 0, stream>>>(semb, sidx, W2, a12, a3, h2T, srcT, s2, l3, r3);
  k_sums<<<N_DIM, 256, 0, stream>>>(adjI, adjA, sidx, s2, r1, l3, r3, invI, invC);
  k_att_inter<<<dim3(R_DIM / 256, N_DIM / 16), 256, 0, stream>>>(adjI, sidx, s2, r1, invI, attI, attT);
  k_att_intra<<<dim3(N_DIM / 256, N_DIM / 16), 256, 0, stream>>>(adjA, sidx, l3, r3, invC, attA);
  k_upre<<<N_DIM / 16, 128, 0, stream>>>(attI, attA, h1T, h2T, u_pre);
  k_t<<<R_DIM / 16, 128, 0, stream>>>(attT, srcT, tmat);
  k_vpre<<<R_DIM, F_DIM, 0, stream>>>(tmat, W1, v_pre);
  k_bn<<<F_DIM, 256, 0, stream>>>(u_pre, u_bf, N_DIM);
  k_bn<<<F_DIM, 256, 0, stream>>>(v_pre, v_bf, R_DIM);
  k_out<<<dim3(N_DIM / 16, R_DIM / 64), 128, 0, stream>>>(u_bf, v_bf, out);
}
